// DCRNN_AE_Model_66305705115739
// MI455X (gfx1250) — matrix operand order repaired and checked by reading, not compiled, not run on hardware
//
#include <hip/hip_runtime.h>
#include <hip/hip_bf16.h>
#include <math.h>

#define NV 8192
#define BATCH 2
#define TT 24
#define HID 32
#define CCOL 33              // per-batch channels in diffusion input (F=1 + H=32)
#define COLS_USED 66         // BATCH * CCOL
#define COLS_PAD 80          // 5 WMMA N-tiles of 16
#define PAD_ELEMS ((COLS_PAD - COLS_USED) * NV)
#define KP 1024              // K-panel staged in LDS (80*KP*2 = 160 KB per buffer)
#define NPANEL (NV / KP)     // 8

typedef __attribute__((ext_vector_type(16))) __bf16        v16bf;
typedef __attribute__((ext_vector_type(8)))  float         v8f;
typedef __attribute__((ext_vector_type(4))) unsigned int   u32x4;
typedef __attribute__((ext_vector_type(8)))  int           i32x8;
typedef __attribute__((ext_vector_type(4)))  int           i32x4;

union Frag {
    struct { u32x4 lo, hi; } q;
    v16bf v;
};

struct BSet { Frag f[5]; };

__device__ __forceinline__ unsigned short f2bf(float f) {
    unsigned int u = __float_as_uint(f);
    u += 0x7FFFu + ((u >> 16) & 1u);       // round to nearest even
    return (unsigned short)(u >> 16);
}

// ---------------------------------------------------------------------------
// TDM: async-load one 1024(K) x 80(col) bf16 panel of the column-major X
// into LDS at byte offset ldsByteOff.  D# per CDNA5 ISA.
//   group0: count=1 | lds_addr | global_addr(57b) | type=2
//   group1: data_size=1(2B) | tensor_dim0=8192 | tensor_dim1=80 |
//           tile_dim0=KP | tile_dim1=80 | tensor_dim0_stride=8192
//   groups 2/3 (+spare group): zero (2-D tensor)
// 6-arg toolchain: (u32x4 g0, i32x8 g1, i32x4 g2, i32x4 g3, i32x8 sp, i32 cpol)
// ---------------------------------------------------------------------------
__device__ __forceinline__ void tdm_load_panel(const unsigned short* gsrc,
                                               unsigned int ldsByteOff) {
    unsigned long long ga = (unsigned long long)(uintptr_t)gsrc;
    u32x4 g0;
    g0[0] = 1u;                                   // count=1 (valid descriptor)
    g0[1] = ldsByteOff;                           // lds_addr (bytes)
    g0[2] = (unsigned int)(ga & 0xFFFFFFFFu);     // global_addr[31:0]
    g0[3] = (unsigned int)((ga >> 32) & 0x1FFFFFFu) | (2u << 30); // [56:32]|type=2
    i32x8 g1;
    g1[0] = (int)(1u << 16);                      // workgroup_mask=0, data_size=1 (2B)
    g1[1] = (int)((8192u & 0xFFFFu) << 16);       // tensor_dim0 lo16 @ bits 63:48
    g1[2] = (int)((8192u >> 16) | (80u << 16));   // tensor_dim0 hi16 | tensor_dim1 lo16
    g1[3] = (int)(((unsigned)KP) << 16);          // tensor_dim1 hi16=0 | tile_dim0=KP
    g1[4] = (int)80u;                             // tile_dim1=80, tile_dim2=0
    g1[5] = (int)8192u;                           // tensor_dim0_stride lo32
    g1[6] = 0;                                    // stride0 hi16 | stride1 lo16
    g1[7] = 0;                                    // stride1 hi32
    i32x4 z4;
    z4[0] = 0; z4[1] = 0; z4[2] = 0; z4[3] = 0;
    i32x8 z8;
    z8[0] = 0; z8[1] = 0; z8[2] = 0; z8[3] = 0;
    z8[4] = 0; z8[5] = 0; z8[6] = 0; z8[7] = 0;
    __builtin_amdgcn_tensor_load_to_lds(g0, g1, z4, z4, z8, 0);
}

// ---------------------------------------------------------------------------
// gso fp32 -> bf16 (row-major). 4 elems / thread.
// ---------------------------------------------------------------------------
__global__ __launch_bounds__(256) void cvt_gso(const float* __restrict__ g,
                                               unsigned short* __restrict__ s) {
    size_t idx = ((size_t)blockIdx.x * 256 + threadIdx.x) * 4;
    float4 v = *(const float4*)(g + idx);
    s[idx + 0] = f2bf(v.x);
    s[idx + 1] = f2bf(v.y);
    s[idx + 2] = f2bf(v.z);
    s[idx + 3] = f2bf(v.w);
}

// ---------------------------------------------------------------------------
// zero hidden state + decoder feedback
// ---------------------------------------------------------------------------
__global__ __launch_bounds__(256) void zero_state(float* __restrict__ h,
                                                  float* __restrict__ prev) {
    int idx = blockIdx.x * 256 + threadIdx.x;     // BATCH*NV*HID threads
    h[idx] = 0.0f;
    if (idx < BATCH * NV) prev[idx] = 0.0f;
}

// ---------------------------------------------------------------------------
// Diffusion hop: Y[80 x NV (col-major)] = S[NV x NV] * X[NV x 80]
//  - S (bf16, row-major) streamed from global/L2, 2-deep register pipeline.
//  - X staged in LDS by the Tensor Data Mover, double-buffered 160KB panels;
//    wave 0 issues tensor_load_to_lds for panel p+1 while the WG computes p.
//  - B fragments double-buffered in registers across K-chunks so ds_loads
//    never overwrite VGPRs an in-flight WMMA is still reading (no WAR nops,
//    no s_wait_dscnt 0 right before the WMMA pack).
//  - One wave: 16 output rows x 80 cols (5 tiles). 8 waves = 128 rows/WG.
// ---------------------------------------------------------------------------
__device__ __forceinline__ void load_bset(BSet& bs, const unsigned short* bufB,
                                          int c, int l15, int bOff) {
#pragma unroll
    for (int t = 0; t < 5; ++t) {
        const unsigned short* bp = bufB + (size_t)(t * 16 + l15) * KP + c + bOff;
        bs.f[t].q.lo = *(const u32x4*)(bp);
        bs.f[t].q.hi = *(const u32x4*)(bp + 8);
    }
}

__global__ __launch_bounds__(256) void diff_wmma(
    const unsigned short* __restrict__ S,
    const unsigned short* __restrict__ Xt,
    float* __restrict__ Yf,
    unsigned short* __restrict__ Yt) {
    extern __shared__ unsigned short ldsX[];      // 2 * 80 * KP bf16 = 320 KB

    const int lane   = threadIdx.x & 31;
    const int wave   = threadIdx.x >> 5;
    const int l15    = lane & 15;
    const int hiHalf = (lane >= 16);
    const int rowBase = blockIdx.x * 128 + wave * 16;

    // A fragment (16x32 bf16): lane<16 -> K0..15 ; lane>=16 -> K16..31
    const unsigned short* aRow = S + (size_t)(rowBase + l15) * NV;
    const int aOff = hiHalf ? 16 : 0;
    // B fragment (32x16 bf16): lane<16 -> col=l15, K0..15 ; lane>=16 -> K16..31
    const int bOff = hiHalf ? 16 : 0;

    v8f acc[5];
#pragma unroll
    for (int t = 0; t < 5; ++t)
#pragma unroll
        for (int j = 0; j < 8; ++j) acc[t][j] = 0.0f;

    // kick off panel 0 (wave 0 only; TDM ignores EXEC, tracked by TENSORcnt)
    if (wave == 0) tdm_load_panel(Xt, 0u);

    // prime the A register pipeline with chunk 0
    Frag aCur;
    aCur.q.lo = *(const u32x4*)(aRow + aOff);
    aCur.q.hi = *(const u32x4*)(aRow + 8 + aOff);

    for (int p = 0; p < NPANEL; ++p) {
        if (wave == 0) {
            if (p + 1 < NPANEL) {
                tdm_load_panel(Xt + (size_t)(p + 1) * KP,
                               (unsigned)(((p + 1) & 1) * (COLS_PAD * KP * 2)));
                __builtin_amdgcn_s_wait_tensorcnt(1);   // panel p landed
            } else {
                __builtin_amdgcn_s_wait_tensorcnt(0);
            }
        }
        __syncthreads();                                 // panel p visible to WG

        const unsigned short* bufB = ldsX + (size_t)(p & 1) * (COLS_PAD * KP);
        const int kbase = p * KP;

        BSet bCur, bNxt;
        load_bset(bCur, bufB, 0, l15, bOff);             // prime B for chunk 0

#pragma unroll 2
        for (int c = 0; c < KP; c += 32) {
            // prefetch next chunk's B fragments into the alternate register set
            int cn = (c + 32 < KP) ? (c + 32) : 0;
            load_bset(bNxt, bufB, cn, l15, bOff);
            // next A fragment (global) + stream-ahead prefetch
            int kn = kbase + c + 32;
            kn = (kn < NV) ? kn : 0;
            __builtin_prefetch(aRow + ((kbase + c + 2048 < NV) ? (kbase + c + 2048)
                                                               : (kbase + c)), 0, 1);
            Frag aNxt;
            aNxt.q.lo = *(const u32x4*)(aRow + kn + aOff);
            aNxt.q.hi = *(const u32x4*)(aRow + kn + 8 + aOff);
#pragma unroll
            for (int t = 0; t < 5; ++t) {
                acc[t] = __builtin_amdgcn_wmma_f32_16x16x32_bf16(
                    false, aCur.v, false, bCur.f[t].v, (short)0, acc[t], false, false);
            }
            aCur = aNxt;
            bCur = bNxt;
        }
        __syncthreads();                                 // WG done with panel p
    }

    // C/D layout: VGPR j, lanes 0-15 -> M=j, lanes 16-31 -> M=j+8 ; N = lane&15
    const int rBase = rowBase + (hiHalf ? 8 : 0);
#pragma unroll
    for (int t = 0; t < 5; ++t) {
        const size_t cb = (size_t)(t * 16 + l15) * NV;
#pragma unroll
        for (int j = 0; j < 8; ++j) {
            float v = acc[t][j];
            size_t off = cb + (size_t)(rBase + j);
            Yf[off] = v;
            Yt[off] = f2bf(v);
        }
    }
}

// ---------------------------------------------------------------------------
// Build XH = [x_t, h] into column-major fp32 + bf16 buffers, zero pad cols.
// Threads: BATCH*NV*HID.  xp[b*xsB + n] is the x feature.
// ---------------------------------------------------------------------------
__global__ __launch_bounds__(256) void pack_xh(
    const float* __restrict__ h,
    const float* __restrict__ xp, int xsB,
    float* __restrict__ XHf, unsigned short* __restrict__ XHt) {
    int idx = blockIdx.x * 256 + threadIdx.x;
    int o = idx & 31;
    int n = (idx >> 5) & (NV - 1);
    int b = idx >> 18;
    float v = h[idx];
    size_t cidx = (size_t)(b * CCOL + 1 + o) * NV + n;
    XHf[cidx] = v;
    XHt[cidx] = f2bf(v);
    if (o == 0) {
        float xv = xp[(size_t)b * xsB + n];
        size_t c0 = (size_t)(b * CCOL) * NV + n;
        XHf[c0] = xv;
        XHt[c0] = f2bf(xv);
    }
    if (idx < PAD_ELEMS) XHt[(size_t)COLS_USED * NV + idx] = 0;
}

// ---------------------------------------------------------------------------
// r,u gates (sigmoid of 99->32 linear over [XH, Z1, Z2]); builds XC=[x, r*h].
// ---------------------------------------------------------------------------
__global__ __launch_bounds__(256) void gates_kernel(
    const float* __restrict__ XHf, const float* __restrict__ Z1f,
    const float* __restrict__ Z2f,
    const float* __restrict__ Wr, const float* __restrict__ br,
    const float* __restrict__ Wu, const float* __restrict__ bu,
    const float* __restrict__ h,
    const float* __restrict__ xp, int xsB,
    float* __restrict__ u_buf,
    float* __restrict__ XCf, unsigned short* __restrict__ XCt) {
    int idx = blockIdx.x * 256 + threadIdx.x;
    int o = idx & 31;
    int n = (idx >> 5) & (NV - 1);
    int b = idx >> 18;
    float accR = br[o], accU = bu[o];
    const float* bufs[3] = {XHf, Z1f, Z2f};
#pragma unroll
    for (int hop = 0; hop < 3; ++hop) {
        const float* bp = bufs[hop] + (size_t)(b * CCOL) * NV + n;
        const float* wr = Wr + (hop * CCOL) * HID + o;
        const float* wu = Wu + (hop * CCOL) * HID + o;
        for (int c = 0; c < CCOL; ++c) {
            float v = bp[(size_t)c * NV];
            accR += v * wr[c * HID];
            accU += v * wu[c * HID];
        }
    }
    float r = 1.0f / (1.0f + expf(-accR));
    float u = 1.0f / (1.0f + expf(-accU));
    u_buf[idx] = u;
    float xc = r * h[idx];
    size_t cidx = (size_t)(b * CCOL + 1 + o) * NV + n;
    XCf[cidx] = xc;
    XCt[cidx] = f2bf(xc);
    if (o == 0) {
        float xv = xp[(size_t)b * xsB + n];
        size_t c0 = (size_t)(b * CCOL) * NV + n;
        XCf[c0] = xv;
        XCt[c0] = f2bf(xv);
    }
    if (idx < PAD_ELEMS) XCt[(size_t)COLS_USED * NV + idx] = 0;
}

// ---------------------------------------------------------------------------
// c = tanh(99->32 linear over [XC, C1, C2]); h = u*h + (1-u)*c (in place)
// ---------------------------------------------------------------------------
__global__ __launch_bounds__(256) void cell_final(
    const float* __restrict__ XCf, const float* __restrict__ C1f,
    const float* __restrict__ C2f,
    const float* __restrict__ Wc, const float* __restrict__ bc,
    const float* __restrict__ u_buf, float* __restrict__ h) {
    int idx = blockIdx.x * 256 + threadIdx.x;
    int o = idx & 31;
    int n = (idx >> 5) & (NV - 1);
    int b = idx >> 18;
    float acc = bc[o];
    const float* bufs[3] = {XCf, C1f, C2f};
#pragma unroll
    for (int hop = 0; hop < 3; ++hop) {
        const float* bp = bufs[hop] + (size_t)(b * CCOL) * NV + n;
        const float* wc = Wc + (hop * CCOL) * HID + o;
        for (int c = 0; c < CCOL; ++c)
            acc += bp[(size_t)c * NV] * wc[c * HID];
    }
    float cv = tanhf(acc);
    float u = u_buf[idx];
    h[idx] = u * h[idx] + (1.0f - u) * cv;
}

// ---------------------------------------------------------------------------
// Decoder head: pred = relu(h@W1+b1)@W2+b2 ; write d_out[b,t,n,0] and prev.
// Threads: BATCH*NV.
// ---------------------------------------------------------------------------
__global__ __launch_bounds__(256) void proj_kernel(
    const float* __restrict__ h,
    const float* __restrict__ W1, const float* __restrict__ b1,
    const float* __restrict__ W2, const float* __restrict__ b2,
    float* __restrict__ out, int t, float* __restrict__ prev) {
    int idx = blockIdx.x * 256 + threadIdx.x;
    int n = idx & (NV - 1);
    int b = idx >> 13;
    const float* hp = h + (size_t)idx * HID;
    float hv[HID];
#pragma unroll
    for (int k = 0; k < HID; ++k) hv[k] = hp[k];
    float pred = b2[0];
    for (int j = 0; j < HID; ++j) {
        float s = b1[j];
#pragma unroll
        for (int k = 0; k < HID; ++k) s += hv[k] * W1[k * HID + j];
        pred += fmaxf(s, 0.0f) * W2[j];
    }
    out[(size_t)(b * TT + t) * NV + n] = pred;
    prev[idx] = pred;
}

// ---------------------------------------------------------------------------
extern "C" void kernel_launch(void* const* d_in, const int* in_sizes, int n_in,
                              void* d_out, int out_size, void* d_ws, size_t ws_size,
                              hipStream_t stream) {
    (void)in_sizes; (void)n_in; (void)out_size; (void)ws_size;
    const float* x      = (const float*)d_in[0];
    // d_in[1] = edge_index (unused; gso already built)
    const float* gso    = (const float*)d_in[2];
    const float* enc_Wr = (const float*)d_in[3];  const float* enc_br = (const float*)d_in[4];
    const float* enc_Wu = (const float*)d_in[5];  const float* enc_bu = (const float*)d_in[6];
    const float* enc_Wc = (const float*)d_in[7];  const float* enc_bc = (const float*)d_in[8];
    const float* dec_Wr = (const float*)d_in[9];  const float* dec_br = (const float*)d_in[10];
    const float* dec_Wu = (const float*)d_in[11]; const float* dec_bu = (const float*)d_in[12];
    const float* dec_Wc = (const float*)d_in[13]; const float* dec_bc = (const float*)d_in[14];
    const float* po_W1  = (const float*)d_in[15]; const float* po_b1  = (const float*)d_in[16];
    const float* po_W2  = (const float*)d_in[17]; const float* po_b2  = (const float*)d_in[18];
    float* dout = (float*)d_out;

    // --- workspace carve-out ---
    char* p = (char*)d_ws;
    unsigned short* Sb = (unsigned short*)p;  p += (size_t)NV * NV * 2;          // 128 MB
    const size_t CB = (size_t)COLS_PAD * NV;
    unsigned short* XHt = (unsigned short*)p; p += CB * 2;
    unsigned short* Z1t = (unsigned short*)p; p += CB * 2;
    unsigned short* Z2t = (unsigned short*)p; p += CB * 2;
    unsigned short* XCt = (unsigned short*)p; p += CB * 2;
    unsigned short* C1t = (unsigned short*)p; p += CB * 2;
    unsigned short* C2t = (unsigned short*)p; p += CB * 2;
    float* XHf = (float*)p; p += CB * 4;
    float* Z1f = (float*)p; p += CB * 4;
    float* Z2f = (float*)p; p += CB * 4;
    float* XCf = (float*)p; p += CB * 4;
    float* C1f = (float*)p; p += CB * 4;
    float* C2f = (float*)p; p += CB * 4;
    float* u_buf = (float*)p; p += (size_t)BATCH * NV * HID * 4;
    float* hbuf  = (float*)p; p += (size_t)BATCH * NV * HID * 4;
    float* prev  = (float*)p; p += (size_t)BATCH * NV * 4;

    const int elemBlocks = (BATCH * NV * HID) / 256;   // 2048
    const int diffBlocks = NV / 128;                   // 64
    const int projBlocks = (BATCH * NV) / 256;         // 64
    const size_t ldsBytes = (size_t)2 * COLS_PAD * KP * 2;   // 320 KB

    cvt_gso<<<(NV * (size_t)NV) / (4 * 256), 256, 0, stream>>>(gso, Sb);
    zero_state<<<elemBlocks, 256, 0, stream>>>(hbuf, prev);

    // ---------------- encoder ----------------
    for (int t = 0; t < TT; ++t) {
        const float* xp = x + (size_t)t * NV;   // batch stride TT*NV
        pack_xh<<<elemBlocks, 256, 0, stream>>>(hbuf, xp, TT * NV, XHf, XHt);
        diff_wmma<<<diffBlocks, 256, ldsBytes, stream>>>(Sb, XHt, Z1f, Z1t);
        diff_wmma<<<diffBlocks, 256, ldsBytes, stream>>>(Sb, Z1t, Z2f, Z2t);
        gates_kernel<<<elemBlocks, 256, 0, stream>>>(XHf, Z1f, Z2f,
            enc_Wr, enc_br, enc_Wu, enc_bu, hbuf, xp, TT * NV, u_buf, XCf, XCt);
        diff_wmma<<<diffBlocks, 256, ldsBytes, stream>>>(Sb, XCt, C1f, C1t);
        diff_wmma<<<diffBlocks, 256, ldsBytes, stream>>>(Sb, C1t, C2f, C2t);
        cell_final<<<elemBlocks, 256, 0, stream>>>(XCf, C1f, C2f,
            enc_Wc, enc_bc, u_buf, hbuf);
    }

    // ---------------- decoder ----------------
    for (int t = 0; t < TT; ++t) {
        pack_xh<<<elemBlocks, 256, 0, stream>>>(hbuf, prev, NV, XHf, XHt);
        diff_wmma<<<diffBlocks, 256, ldsBytes, stream>>>(Sb, XHt, Z1f, Z1t);
        diff_wmma<<<diffBlocks, 256, ldsBytes, stream>>>(Sb, Z1t, Z2f, Z2t);
        gates_kernel<<<elemBlocks, 256, 0, stream>>>(XHf, Z1f, Z2f,
            dec_Wr, dec_br, dec_Wu, dec_bu, hbuf, prev, NV, u_buf, XCf, XCt);
        diff_wmma<<<diffBlocks, 256, ldsBytes, stream>>>(Sb, XCt, C1f, C1t);
        diff_wmma<<<diffBlocks, 256, ldsBytes, stream>>>(Sb, C1t, C2f, C2t);
        cell_final<<<elemBlocks, 256, 0, stream>>>(XCf, C1f, C2f,
            dec_Wc, dec_bc, u_buf, hbuf);
        proj_kernel<<<projBlocks, 256, 0, stream>>>(hbuf, po_W1, po_b1,
            po_W2, po_b2, dout, t, prev);
    }
}
